// BasicRNN_61701500174755
// MI455X (gfx1250) — matrix operand order repaired and checked by reading, not compiled, not run on hardware
//
#include <hip/hip_runtime.h>

// ---------------------------------------------------------------------------
// BasicRNN on MI455X (gfx1250): bf16 WMMA recurrence + projection.
//   s_t = tanh([s_{t-1} | x_t] @ [Wh ; Wx] + b)   (fused, K = 1024+256 = 1280)
//   out = states @ Wout + bout
// All GEMMs via v_wmma_f32_16x16x32_bf16 (wave32, 16x16 tiles, f32 accum).
// ---------------------------------------------------------------------------

static constexpr int Bn = 64, Fn = 256, Tn = 1024, Hn = 1024, On = 512;
static constexpr int KT_WH  = Hn / 32;   // 32 K-blocks from Wh
static constexpr int KT_WX  = Fn / 32;   // 8  K-blocks from Wx
static constexpr int KT_CAT = KT_WH + KT_WX; // 40
static constexpr int KT_OUT = Hn / 32;   // 32

typedef __attribute__((ext_vector_type(16))) __bf16 v16bf;
typedef __attribute__((ext_vector_type(8)))  float  v8f;

struct U32x8 { uint4 lo, hi; };

// A-fragment (16x32 bf16, row-major source): lane l covers row (l&15),
// K = (l>>4)*8 + {0..7} in VGPRs 0-3 and +16 in VGPRs 4-7  -> two b128 loads.
__device__ __forceinline__ v16bf load_frag_a(const __bf16* p) {
  U32x8 r;
  r.lo = *(const uint4*)(p);
  r.hi = *(const uint4*)(p + 16);
  return __builtin_bit_cast(v16bf, r);
}

// B-fragment from pre-packed layout: 32 contiguous bytes per lane per K-block.
__device__ __forceinline__ v16bf load_frag_b(const unsigned int* p) {
  U32x8 r;
  r.lo = *(const uint4*)(p);
  r.hi = *(const uint4*)(p + 4);
  return __builtin_bit_cast(v16bf, r);
}

__device__ __forceinline__ v8f wmma_bf16(v16bf a, v16bf b, v8f c) {
  // (neg_a, A, neg_b, B, c_mod, C, reuse_a, reuse_b)
  return __builtin_amdgcn_wmma_f32_16x16x32_bf16(false, a, false, b,
                                                 (short)0, c, false, false);
}

__device__ __forceinline__ unsigned int pack2bf(float lo, float hi) {
  unsigned short a = __builtin_bit_cast(unsigned short, (__bf16)lo);
  unsigned short b = __builtin_bit_cast(unsigned short, (__bf16)hi);
  return (unsigned int)a | ((unsigned int)b << 16);
}

// ---------------------------------------------------------------------------
// Prep kernels
// ---------------------------------------------------------------------------

// Zero initial hidden state (row 0 of the state ring), 64*1024 bf16.
__global__ void zero_s0_kernel(unsigned int* s0) {
  int id = blockIdx.x * blockDim.x + threadIdx.x;
  if (id < (Bn * Hn) / 2) s0[id] = 0u;
}

// x (B,F,T) f32 -> xbf (T,B,F) bf16, tiled LDS transpose (coalesced both sides).
__global__ __launch_bounds__(256) void transpose_x_kernel(
    const float* __restrict__ x, __bf16* __restrict__ xbf) {
  __shared__ float tile[32][33];
  const int tx = threadIdx.x;          // 0..31
  const int ty = threadIdx.y;          // 0..7
  const int t0 = blockIdx.x * 32;
  const int f0 = blockIdx.y * 32;
  const int b  = blockIdx.z;
#pragma unroll
  for (int i = 0; i < 4; ++i) {
    int f = f0 + ty + i * 8;
    tile[ty + i * 8][tx] = x[((size_t)b * Fn + f) * Tn + t0 + tx];
  }
  __syncthreads();
#pragma unroll
  for (int i = 0; i < 4; ++i) {
    int t = t0 + ty + i * 8;
    xbf[((size_t)t * Bn + b) * Fn + f0 + tx] = (__bf16)tile[tx][ty + i * 8];
  }
}

// Pack Wcat = [Wh(1024,H) ; Wx(256,H)] into WMMA B-fragment layout:
// dword index = ((nt*KT_CAT + kt)*32 + lane)*8 + j
// lane covers n = nt*16 + (lane&15); dword j holds K = kt*32+(lane>>4)*8+(j>>2)*16+2*(j&3), +1.
__global__ void pack_wcat_kernel(const float* __restrict__ Wx,
                                 const float* __restrict__ Wh,
                                 unsigned int* __restrict__ dst) {
  int id   = blockIdx.x * blockDim.x + threadIdx.x;  // 64*40*32*8 = 655360
  int j    = id & 7;
  int l    = (id >> 3) & 31;
  int ktnt = id >> 8;
  int kt   = ktnt % KT_CAT;
  int n    = (ktnt / KT_CAT) * 16 + (l & 15);
  int k    = kt * 32 + (l >> 4) * 8 + (j >> 2) * 16 + 2 * (j & 3);
  float lo, hi;
  if (k < Hn) { lo = Wh[(size_t)k * Hn + n];        hi = Wh[(size_t)(k + 1) * Hn + n]; }
  else        { lo = Wx[(size_t)(k - Hn) * Hn + n]; hi = Wx[(size_t)(k + 1 - Hn) * Hn + n]; }
  dst[id] = pack2bf(lo, hi);
}

// Pack Wout (H,O) the same way: [nt=32][kt=32][lane][j].
__global__ void pack_wout_kernel(const float* __restrict__ Wout,
                                 unsigned int* __restrict__ dst) {
  int id   = blockIdx.x * blockDim.x + threadIdx.x;  // 32*32*32*8 = 262144
  int j    = id & 7;
  int l    = (id >> 3) & 31;
  int ktnt = id >> 8;
  int kt   = ktnt % KT_OUT;
  int n    = (ktnt / KT_OUT) * 16 + (l & 15);
  int k    = kt * 32 + (l >> 4) * 8 + (j >> 2) * 16 + 2 * (j & 3);
  dst[id] = pack2bf(Wout[(size_t)k * On + n], Wout[(size_t)(k + 1) * On + n]);
}

// ---------------------------------------------------------------------------
// Recurrent step: S[t+1] = tanh([S[t] | x_t] @ Wcat + b), states kept bf16.
// 256 output tiles (4 M-tiles x 64 N-tiles), one wave per tile, 40 WMMA/wave.
// ---------------------------------------------------------------------------
__global__ __launch_bounds__(256) void rnn_step_kernel(
    __bf16* __restrict__ S,               // (T+1, B, H) bf16 state ring
    const __bf16* __restrict__ xbf,       // (T, B, F) bf16
    const unsigned int* __restrict__ Wp,  // packed Wcat
    const float* __restrict__ bias, int t) {
  const int lane  = threadIdx.x & 31;
  const int wave  = threadIdx.x >> 5;
  const int tile  = blockIdx.x * 8 + wave;   // 0..255
  const int mtile = tile >> 6;               // 0..3
  const int ntile = tile & 63;               // 0..63
  const int mrow  = lane & 15;
  const int khalf = lane >> 4;

  const __bf16* arow = S + (size_t)t * (Bn * Hn)
                         + (size_t)(mtile * 16 + mrow) * Hn + khalf * 8;
  const __bf16* xrow = xbf + ((size_t)t * Bn + mtile * 16 + mrow) * Fn + khalf * 8;
  const unsigned int* bbase = Wp + ((size_t)ntile * KT_CAT * 32 + lane) * 8;

  v8f acc0 = {}; v8f acc1 = {};   // two accumulators hide WMMA RAW latency
#pragma unroll 4
  for (int kt = 0; kt < KT_WH; ++kt) {
    v16bf a = load_frag_a(arow + kt * 32);
    v16bf b = load_frag_b(bbase + kt * 256);
    __builtin_prefetch(bbase + (kt + 2) * 256, 0, 1);  // global_prefetch_b8
    if (kt & 1) acc1 = wmma_bf16(a, b, acc1); else acc0 = wmma_bf16(a, b, acc0);
  }
#pragma unroll
  for (int kt = 0; kt < KT_WX; ++kt) {
    v16bf a = load_frag_a(xrow + kt * 32);
    v16bf b = load_frag_b(bbase + (KT_WH + kt) * 256);
    if (kt & 1) acc1 = wmma_bf16(a, b, acc1); else acc0 = wmma_bf16(a, b, acc0);
  }

  const int n  = ntile * 16 + (lane & 15);
  const float bv = bias[n];
  __bf16* dst = S + (size_t)(t + 1) * (Bn * Hn);
#pragma unroll
  for (int i = 0; i < 8; ++i) {
    int row = mtile * 16 + i + 8 * khalf;   // C/D layout: M = i (+8 for hi lanes)
    float v = tanhf(acc0[i] + acc1[i] + bv);
    dst[(size_t)row * Hn + n] = (__bf16)v;
  }
}

// ---------------------------------------------------------------------------
// Projection: out(B*T, O) = states(B*T, H) @ Wout + bout   (f32 output)
// 4096 M-tiles x 32 N-tiles, one wave per tile, 32 WMMA/wave.
// ---------------------------------------------------------------------------
__global__ __launch_bounds__(256) void proj_kernel(
    const __bf16* __restrict__ S, const unsigned int* __restrict__ Wp,
    const float* __restrict__ bout, float* __restrict__ out) {
  const int lane  = threadIdx.x & 31;
  const int wave  = threadIdx.x >> 5;
  const int gwave = blockIdx.x * 8 + wave;   // 0..131071
  const int mt    = gwave >> 5;              // 0..4095
  const int ntile = gwave & 31;              // 0..31
  const int khalf = lane >> 4;

  const int r0 = mt * 16 + (lane & 15);      // global row = b*T + t
  const int b  = r0 >> 10;                   // T = 1024
  const int t  = r0 & 1023;
  const __bf16* arow = S + ((size_t)(t + 1) * Bn + b) * Hn + khalf * 8;
  const unsigned int* bbase = Wp + ((size_t)ntile * KT_OUT * 32 + lane) * 8;

  v8f acc0 = {}; v8f acc1 = {};
#pragma unroll 4
  for (int kt = 0; kt < KT_OUT; ++kt) {
    v16bf a  = load_frag_a(arow + kt * 32);
    v16bf bb = load_frag_b(bbase + kt * 256);
    __builtin_prefetch(bbase + (kt + 2) * 256, 0, 1);
    if (kt & 1) acc1 = wmma_bf16(a, bb, acc1); else acc0 = wmma_bf16(a, bb, acc0);
  }
  const int n  = ntile * 16 + (lane & 15);
  const float bo = bout[n];
#pragma unroll
  for (int i = 0; i < 8; ++i) {
    int r = mt * 16 + i + 8 * khalf;
    out[(size_t)r * On + n] = acc0[i] + acc1[i] + bo;
  }
}

// ---------------------------------------------------------------------------
extern "C" void kernel_launch(void* const* d_in, const int* in_sizes, int n_in,
                              void* d_out, int out_size, void* d_ws, size_t ws_size,
                              hipStream_t stream) {
  (void)in_sizes; (void)n_in; (void)out_size; (void)ws_size;
  const float* x    = (const float*)d_in[0];   // (B,F,T)
  const float* Wx   = (const float*)d_in[1];   // (F,H)
  const float* Wh   = (const float*)d_in[2];   // (H,H)
  const float* bias = (const float*)d_in[3];   // (H,)
  const float* Wout = (const float*)d_in[4];   // (H,O)
  const float* bout = (const float*)d_in[5];   // (O,)
  float* out = (float*)d_out;                  // (B,T,O) f32

  char* ws = (char*)d_ws;
  size_t off = 0;
  auto carve = [&](size_t bytes) {
    void* p = ws + off;
    off += (bytes + 255) & ~(size_t)255;
    return p;
  };
  __bf16*       Sbuf  = (__bf16*)carve((size_t)(Tn + 1) * Bn * Hn * 2); // 134.3 MB
  __bf16*       xbf   = (__bf16*)carve((size_t)Tn * Bn * Fn * 2);       //  33.6 MB
  unsigned int* Wcatp = (unsigned int*)carve((size_t)(Hn + Fn) * Hn * 2); // 2.6 MB
  unsigned int* Woutp = (unsigned int*)carve((size_t)Hn * On * 2);        // 1.0 MB

  zero_s0_kernel<<<128, 256, 0, stream>>>((unsigned int*)Sbuf);
  transpose_x_kernel<<<dim3(Tn / 32, Fn / 32, Bn), dim3(32, 8), 0, stream>>>(x, xbf);
  pack_wcat_kernel<<<(64 * KT_CAT * 32 * 8) / 256, 256, 0, stream>>>(Wx, Wh, Wcatp);
  pack_wout_kernel<<<(32 * KT_OUT * 32 * 8) / 256, 256, 0, stream>>>(Wout, Woutp);

  for (int t = 0; t < Tn; ++t)
    rnn_step_kernel<<<32, 256, 0, stream>>>(Sbuf, xbf, Wcatp, bias, t);

  proj_kernel<<<(4096 * 32) / 8, 256, 0, stream>>>(Sbuf, Woutp, bout, out);
}
